// SessionGraph4_48000554500648
// MI455X (gfx1250) — compile-verified, matrix operand order checked by reading, not run on hardware
//
#include <hip/hip_runtime.h>
#include <math.h>

static constexpr int kV = 100000;
static constexpr int kD = 256;
static constexpr int kB = 512;
static constexpr int kS = 20;
static constexpr int kN = kB * kS;   // 10240
static constexpr int kE = 8 * kN;    // 81920
static constexpr float kSCALE = 12.0f;

typedef __attribute__((ext_vector_type(16))) _Float16 v16h;
typedef __attribute__((ext_vector_type(8)))  float    v8f;
typedef _Float16 f16;

// ---- monotonic float<->uint encoding for atomic max on signed floats ----
__device__ __forceinline__ unsigned enc_f32(float f) {
  int i = __float_as_int(f);
  return (i >= 0) ? ((unsigned)i ^ 0x80000000u) : ~(unsigned)i;
}
__device__ __forceinline__ float dec_f32(unsigned u) {
  int i = (u & 0x80000000u) ? (int)(u ^ 0x80000000u) : ~(int)u;
  return __int_as_float(i);
}

// ===================== init / normalize / gather =====================

__global__ void k_init(float* __restrict__ h, float* __restrict__ esum,
                       unsigned* __restrict__ emax) {
  int i = blockIdx.x * 256 + threadIdx.x;
  if (i < kN * kD) h[i] = 0.f;
  if (i < kN) { esum[i] = 0.f; emax[i] = 0u; }
}

// one block (256 thr) per emb row: inv_norm + f16 normalized copy (bn)
__global__ void k_rownorm(const float* __restrict__ emb,
                          float* __restrict__ inv_norm, f16* __restrict__ bn) {
  __shared__ float red[256];
  int row = blockIdx.x, t = threadIdx.x;
  float x = emb[(size_t)row * kD + t];
  red[t] = x * x; __syncthreads();
  for (int off = 128; off; off >>= 1) { if (t < off) red[t] += red[t + off]; __syncthreads(); }
  float inv = 1.f / fmaxf(sqrtf(red[0]), 1e-12f);
  if (t == 0) inv_norm[row] = inv;
  bn[(size_t)row * kD + t] = (f16)(x * inv);
}

// h_v[n] = emb[iid[n]] * inv_norm[iid[n]]   (f32, needed for graph math)
__global__ void k_gather_hv(const float* __restrict__ emb,
                            const float* __restrict__ inv_norm,
                            const int* __restrict__ iid, float* __restrict__ h_v) {
  int n = blockIdx.x, t = threadIdx.x;
  int r = iid[n];
  h_v[(size_t)n * kD + t] = emb[(size_t)r * kD + t] * inv_norm[r];
}

// ===================== GAT edge attention =====================

// one wave32 per edge; 8 dims/lane; shfl reduce; write e + atomicMax(enc)
__global__ void k_edge_scores(const float* __restrict__ h_v,
                              const int* __restrict__ src, const int* __restrict__ dst,
                              const int* __restrict__ dis,
                              const float* __restrict__ dis_emb,
                              const float* __restrict__ pi_w, const float* __restrict__ M_w,
                              float* __restrict__ e_out, unsigned* __restrict__ emax) {
  int wave = threadIdx.x >> 5, lane = threadIdx.x & 31;
  int e = blockIdx.x * 8 + wave;
  const float* hs = h_v + (size_t)src[e] * kD;
  const float* hd = h_v + (size_t)dst[e] * kD;
  const float* hx = dis_emb + (size_t)dis[e] * kD;
  float s1 = 0.f, s2 = 0.f;
#pragma unroll
  for (int j = 0; j < 8; ++j) {
    int d = lane + 32 * j;
    float p = hs[d] * hd[d];
    float x = hx[d];
    s1 += p * x * pi_w[d];
    s2 += p * M_w[d] + x * M_w[kD + d];
  }
  for (int off = 16; off; off >>= 1) {
    s1 += __shfl_xor(s1, off);
    s2 += __shfl_xor(s2, off);
  }
  if (lane == 0) {
    float gate = 1.f / (1.f + __expf(-s2));
    float ev = s1 * gate;
    e_out[e] = ev;
    atomicMax(&emax[dst[e]], enc_f32(ev));
  }
}

__global__ void k_edge_expsum(const float* __restrict__ e_out,
                              const int* __restrict__ dst,
                              const unsigned* __restrict__ emax,
                              float* __restrict__ ee, float* __restrict__ esum) {
  int e = blockIdx.x * 256 + threadIdx.x;
  int dn = dst[e];
  float x = __expf(e_out[e] - dec_f32(emax[dn]));
  ee[e] = x;
  unsafeAtomicAdd(&esum[dn], x);
}

// h[dst] += h_v[src] * a  (one wave per edge)
__global__ void k_edge_aggr(const float* __restrict__ h_v,
                            const int* __restrict__ src, const int* __restrict__ dst,
                            const float* __restrict__ ee, const float* __restrict__ esum,
                            float* __restrict__ h) {
  int wave = threadIdx.x >> 5, lane = threadIdx.x & 31;
  int e = blockIdx.x * 8 + wave;
  int dn = dst[e];
  float a = ee[e] / esum[dn];
  const float* hs = h_v + (size_t)src[e] * kD;
  float* hrow = h + (size_t)dn * kD;
#pragma unroll
  for (int j = 0; j < 8; ++j) {
    int d = lane + 32 * j;
    unsafeAtomicAdd(&hrow[d], hs[d] * a);
  }
}

// ===================== f16 packing for WMMA GEMMs =====================

// h2[n, 0:512] = concat(h[n], pos_emb[n % S])  -> f16
__global__ void k_build_h2(const float* __restrict__ h, const float* __restrict__ pos_emb,
                           f16* __restrict__ h2) {
  int idx = blockIdx.x * 256 + threadIdx.x;       // kN*512 total
  int n = idx >> 9, c = idx & 511;
  float v = (c < kD) ? h[(size_t)n * kD + c]
                     : pos_emb[(size_t)(n % kS) * kD + (c - kD)];
  h2[idx] = (f16)v;
}

// a2[b, 0:512] = concat(target_emb[tid[b]], h[last_idx[b]])  -> f16
__global__ void k_build_a2(const float* __restrict__ h, const float* __restrict__ target_emb,
                           const int* __restrict__ tid, const int* __restrict__ last_idx,
                           f16* __restrict__ a2) {
  int idx = blockIdx.x * 256 + threadIdx.x;       // kB*512 total
  int b = idx >> 9, c = idx & 511;
  float v = (c < kD) ? target_emb[(size_t)tid[b] * kD + c]
                     : h[(size_t)last_idx[b] * kD + (c - kD)];
  a2[idx] = (f16)v;
}

// transpose+convert q_w and r_w ([2D,D] rm) into [D,2D] rm f16
__global__ void k_weightsT(const float* __restrict__ q_w, const float* __restrict__ r_w,
                           f16* __restrict__ qwT, f16* __restrict__ rwT) {
  int idx = blockIdx.x * 256 + threadIdx.x;       // 2 * 512*256 total
  const float* Wsrc = (idx < 2 * kD * kD) ? q_w : r_w;
  f16* Wdst = (idx < 2 * kD * kD) ? qwT : rwT;
  int i = idx & (2 * kD * kD - 1);
  int j = i & (kD - 1);                           // out col (0..255)
  int k = i >> 8;                                 // 0..511
  Wdst[(size_t)j * (2 * kD) + k] = (f16)Wsrc[(size_t)k * kD + j];
}

// ===================== WMMA NT GEMM: C[M,N] = A[M,K] * B[N,K]^T =====================
// block = 256 thr = 8 waves; per-wave 64M x 16N macro-tile (4 accumulators):
// each B fragment is loaded once and multiplied against 4 A fragments, cutting
// B-stream traffic 4x on the dominant logits GEMM (bn_f16 streams from L2).
// A fragment per ISA 16-bit 16x32 layout; B mirrored (N from lane).
// Requires: M % 64 == 0, N % 16 == 0, K % 32 == 0 (true at all call sites).
__global__ void k_gemm_nt(const f16* __restrict__ A, const f16* __restrict__ B,
                          float* __restrict__ C, int M, int N, int K) {
  int wave = threadIdx.x >> 5;
  int lane = threadIdx.x & 31;
  int n0 = (blockIdx.x * 8 + wave) * 16;
  int m0 = blockIdx.y * 64;
  if (n0 >= N) return;                            // whole-wave cull: EXEC stays all-ones
  int half = lane >> 4, lsub = lane & 15;
  int pbase = half * 4;                           // pair index base: (half*8)/2
  const unsigned* bp  = (const unsigned*)(B + (size_t)(n0 + lsub) * K);
  const unsigned* ap0 = (const unsigned*)(A + (size_t)(m0 +  0 + lsub) * K);
  const unsigned* ap1 = (const unsigned*)(A + (size_t)(m0 + 16 + lsub) * K);
  const unsigned* ap2 = (const unsigned*)(A + (size_t)(m0 + 32 + lsub) * K);
  const unsigned* ap3 = (const unsigned*)(A + (size_t)(m0 + 48 + lsub) * K);
  v8f acc0 = {}, acc1 = {}, acc2 = {}, acc3 = {};
  for (int kk = 0; kk < K; kk += 32) {
    int p = (kk >> 1) + pbase;
    union { v16h v; unsigned u[8]; } bf, a0, a1, a2, a3;
#pragma unroll
    for (int i = 0; i < 4; ++i) {
      bf.u[i] = bp[p + i];  bf.u[4 + i] = bp[p + 8 + i];
      a0.u[i] = ap0[p + i]; a0.u[4 + i] = ap0[p + 8 + i];
      a1.u[i] = ap1[p + i]; a1.u[4 + i] = ap1[p + 8 + i];
      a2.u[i] = ap2[p + i]; a2.u[4 + i] = ap2[p + 8 + i];
      a3.u[i] = ap3[p + i]; a3.u[4 + i] = ap3[p + 8 + i];
    }
    if (kk + 32 < K) __builtin_prefetch(bp + p + 16, 0, 3);  // global_prefetch_b8 (near)
    acc0 = __builtin_amdgcn_wmma_f32_16x16x32_f16(false, a0.v, false, bf.v, (short)0, acc0, false, false);
    acc1 = __builtin_amdgcn_wmma_f32_16x16x32_f16(false, a1.v, false, bf.v, (short)0, acc1, false, false);
    acc2 = __builtin_amdgcn_wmma_f32_16x16x32_f16(false, a2.v, false, bf.v, (short)0, acc2, false, false);
    acc3 = __builtin_amdgcn_wmma_f32_16x16x32_f16(false, a3.v, false, bf.v, (short)0, acc3, false, false);
  }
  int n = n0 + lsub;
  int mb = m0 + half * 8;
#pragma unroll
  for (int r = 0; r < 8; ++r) {
    C[(size_t)(mb +  0 + r) * N + n] = acc0[r];
    C[(size_t)(mb + 16 + r) * N + n] = acc1[r];
    C[(size_t)(mb + 32 + r) * N + n] = acc2[r];
    C[(size_t)(mb + 48 + r) * N + n] = acc3[r];
  }
}

// ===================== PosAggregator epilogue + sr l2norm =====================

__global__ void k_posagg(const float* __restrict__ X1, const float* __restrict__ f,
                         const float* __restrict__ h,
                         float* __restrict__ sr, f16* __restrict__ sr16) {
  __shared__ float red[256];
  int b = blockIdx.x, t = threadIdx.x;
  float fd = f[(size_t)b * kD + t];
  float srd = 0.f;
  for (int j = 0; j < kS; ++j) {
    int n = b * kS + j;
    red[t] = tanhf(X1[(size_t)n * kD + t]) * fd;
    __syncthreads();
    for (int off = 128; off; off >>= 1) { if (t < off) red[t] += red[t + off]; __syncthreads(); }
    float dot = red[0];
    __syncthreads();
    srd += h[(size_t)n * kD + t] * dot;
  }
  red[t] = srd * srd; __syncthreads();
  for (int off = 128; off; off >>= 1) { if (t < off) red[t] += red[t + off]; __syncthreads(); }
  float inv = 1.f / fmaxf(sqrtf(red[0]), 1e-12f);
  float v = srd * inv;
  sr[(size_t)b * kD + t] = v;
  sr16[(size_t)b * kD + t] = (f16)v;
}

// ===================== phi head: log softmax(relu(sr@sc1+b)@sc2) =====================

__global__ void k_phi(const float* __restrict__ sr, const float* __restrict__ sc1w,
                      const float* __restrict__ sc1b, const float* __restrict__ sc2w,
                      float* __restrict__ logphi) {
  __shared__ float sh[kD];
  __shared__ float red[256];
  int b = blockIdx.x, t = threadIdx.x;
  sh[t] = sr[(size_t)b * kD + t]; __syncthreads();
  float acc = sc1b[t];
  for (int d = 0; d < kD; ++d) acc += sh[d] * sc1w[(size_t)d * kD + t];
  float y = fmaxf(acc, 0.f);
  red[t] = y * sc2w[t * 2 + 0]; __syncthreads();
  for (int off = 128; off; off >>= 1) { if (t < off) red[t] += red[t + off]; __syncthreads(); }
  float z0 = red[0]; __syncthreads();
  red[t] = y * sc2w[t * 2 + 1]; __syncthreads();
  for (int off = 128; off; off >>= 1) { if (t < off) red[t] += red[t + off]; __syncthreads(); }
  float z1 = red[0];
  if (t == 0) {
    float m = fmaxf(z0, z1);
    float lse = m + logf(__expf(z0 - m) + __expf(z1 - m));
    logphi[b * 2 + 0] = z0 - lse;
    logphi[b * 2 + 1] = z1 - lse;
  }
}

// ===================== finalize: in-place masked log-softmax of d_out =====================
// out[b,v] = 12*l + (masked ? logphi0 - m_in - log s_in : logphi1 - m_ex - log s_ex)

__global__ void k_final(float* __restrict__ out, const int* __restrict__ iid,
                        const float* __restrict__ logphi) {
  __shared__ int sid[kS];
  __shared__ unsigned long long sh_bloom;
  __shared__ float red_m[256], red_s[256];
  __shared__ float sh_cin;
  int b = blockIdx.x, t = threadIdx.x;
  float* row = out + (size_t)b * kV;
  if (t < kS) sid[t] = iid[b * kS + t];
  __syncthreads();
  if (t == 0) {
    unsigned long long bl = 0;
    for (int j = 0; j < kS; ++j)
      bl |= 1ull << (((unsigned)sid[j] * 2654435761u) >> 26);
    sh_bloom = bl;
    float lv[kS]; int uq[kS];
    float mi = -INFINITY;
    for (int j = 0; j < kS; ++j) {
      int u = 1;
      for (int j2 = 0; j2 < j; ++j2) if (sid[j2] == sid[j]) { u = 0; break; }
      uq[j] = u;
      lv[j] = 0.f;
      if (u) { lv[j] = kSCALE * row[sid[j]]; mi = fmaxf(mi, lv[j]); }
    }
    float si = 0.f;
    for (int j = 0; j < kS; ++j) if (uq[j]) si += __expf(lv[j] - mi);
    sh_cin = logphi[b * 2 + 0] - mi - logf(si);
  }
  __syncthreads();
  unsigned long long bl = sh_bloom;
  float cin = sh_cin;
  // pass 1: online softmax stats over unmasked columns
  float m = -INFINITY, s = 0.f;
  for (int v = t; v < kV; v += 256) {
    bool masked = false;
    unsigned bit = ((unsigned)v * 2654435761u) >> 26;
    if ((bl >> bit) & 1ull) {
      for (int j = 0; j < kS; ++j) masked |= (v == sid[j]);
    }
    if (!masked) {
      float l = kSCALE * row[v];
      if (l > m) { s = s * __expf(m - l) + 1.f; m = l; }
      else s += __expf(l - m);
    }
  }
  red_m[t] = m; red_s[t] = s; __syncthreads();
  for (int off = 128; off; off >>= 1) {
    if (t < off) {
      float m1 = red_m[t], s1 = red_s[t], m2 = red_m[t + off], s2 = red_s[t + off];
      float mm = fmaxf(m1, m2), ss = 0.f;
      if (m1 > -INFINITY) ss += s1 * __expf(m1 - mm);
      if (m2 > -INFINITY) ss += s2 * __expf(m2 - mm);
      red_m[t] = mm; red_s[t] = ss;
    }
    __syncthreads();
  }
  float cex = logphi[b * 2 + 1] - red_m[0] - logf(red_s[0]);
  // pass 2: in-place rewrite
  for (int v = t; v < kV; v += 256) {
    bool masked = false;
    unsigned bit = ((unsigned)v * 2654435761u) >> 26;
    if ((bl >> bit) & 1ull) {
      for (int j = 0; j < kS; ++j) masked |= (v == sid[j]);
    }
    float l = kSCALE * row[v];
    row[v] = l + (masked ? cin : cex);
  }
}

// ===================== host launch =====================

extern "C" void kernel_launch(void* const* d_in, const int* in_sizes, int n_in,
                              void* d_out, int out_size, void* d_ws, size_t ws_size,
                              hipStream_t stream) {
  (void)in_sizes; (void)n_in; (void)out_size; (void)ws_size;
  const int*   iid        = (const int*)d_in[0];
  const int*   src        = (const int*)d_in[1];
  const int*   dst        = (const int*)d_in[2];
  const int*   dis        = (const int*)d_in[3];
  const int*   tid        = (const int*)d_in[7];
  const int*   last_idx   = (const int*)d_in[8];
  const float* emb        = (const float*)d_in[9];
  const float* pos_emb    = (const float*)d_in[10];
  const float* dis_emb    = (const float*)d_in[11];
  const float* target_emb = (const float*)d_in[12];
  const float* pi_w       = (const float*)d_in[13];
  const float* M_w        = (const float*)d_in[14];
  const float* q_w        = (const float*)d_in[15];
  const float* r_w        = (const float*)d_in[16];
  const float* sc1_w      = (const float*)d_in[17];
  const float* sc1_b      = (const float*)d_in[18];
  const float* sc2_w      = (const float*)d_in[19];
  float* out = (float*)d_out;

  // ---- workspace bump allocator (256B aligned) ----
  char* w = (char*)d_ws;
  auto carve = [&](size_t bytes) -> void* {
    void* p = (void*)w;
    w += (bytes + 255) & ~(size_t)255;
    return p;
  };
  float*    inv_norm = (float*)   carve((size_t)kV * 4);
  f16*      bn16     = (f16*)     carve((size_t)kV * kD * 2);
  float*    h_v      = (float*)   carve((size_t)kN * kD * 4);
  float*    e_buf    = (float*)   carve((size_t)kE * 4);
  float*    ee_buf   = (float*)   carve((size_t)kE * 4);
  unsigned* emax     = (unsigned*)carve((size_t)kN * 4);
  float*    esum     = (float*)   carve((size_t)kN * 4);
  float*    h_agg    = (float*)   carve((size_t)kN * kD * 4);
  f16*      h2       = (f16*)     carve((size_t)kN * 2 * kD * 2);
  f16*      qwT      = (f16*)     carve((size_t)kD * 2 * kD * 2);
  f16*      rwT      = (f16*)     carve((size_t)kD * 2 * kD * 2);
  float*    X1       = (float*)   carve((size_t)kN * kD * 4);
  f16*      a2       = (f16*)     carve((size_t)kB * 2 * kD * 2);
  float*    f_buf    = (float*)   carve((size_t)kB * kD * 4);
  float*    sr       = (float*)   carve((size_t)kB * kD * 4);
  f16*      sr16     = (f16*)     carve((size_t)kB * kD * 2);
  float*    logphi   = (float*)   carve((size_t)kB * 2 * 4);

  // 1. zero atomics targets
  k_init<<<(kN * kD + 255) / 256, 256, 0, stream>>>(h_agg, esum, emax);
  // 2. normalize embedding table -> inv_norm + bn_f16
  k_rownorm<<<kV, 256, 0, stream>>>(emb, inv_norm, bn16);
  // 3. gather h_v (f32)
  k_gather_hv<<<kN, 256, 0, stream>>>(emb, inv_norm, iid, h_v);
  // 4. edge attention scores + segment max
  k_edge_scores<<<kE / 8, 256, 0, stream>>>(h_v, src, dst, dis, dis_emb, pi_w, M_w,
                                            e_buf, emax);
  // 5. exp + segment sum
  k_edge_expsum<<<kE / 256, 256, 0, stream>>>(e_buf, dst, emax, ee_buf, esum);
  // 6. weighted aggregation into h
  k_edge_aggr<<<kE / 8, 256, 0, stream>>>(h_v, src, dst, ee_buf, esum, h_agg);
  // 7. pack f16 operands
  k_weightsT<<<(2 * 2 * kD * kD + 255) / 256, 256, 0, stream>>>(q_w, r_w, qwT, rwT);
  k_build_h2<<<(kN * 2 * kD + 255) / 256, 256, 0, stream>>>(h_agg, pos_emb, h2);
  k_build_a2<<<(kB * 2 * kD + 255) / 256, 256, 0, stream>>>(h_agg, target_emb, tid,
                                                            last_idx, a2);
  // 8. X1 = h2 @ q_w  (WMMA, M=10240 N=256 K=512)
  {
    dim3 g(((kD / 16) + 7) / 8, kN / 64);
    k_gemm_nt<<<g, 256, 0, stream>>>(h2, qwT, X1, kN, kD, 2 * kD);
  }
  // 9. f = a2 @ r_w  (WMMA, M=512 N=256 K=512)
  {
    dim3 g(((kD / 16) + 7) / 8, kB / 64);
    k_gemm_nt<<<g, 256, 0, stream>>>(a2, rwT, f_buf, kB, kD, 2 * kD);
  }
  // 10. PosAggregator epilogue + sr l2norm
  k_posagg<<<kB, 256, 0, stream>>>(X1, f_buf, h_agg, sr, sr16);
  // 11. phi head
  k_phi<<<kB, 256, 0, stream>>>(sr, sc1_w, sc1_b, sc2_w, logphi);
  // 12. logits = sr @ bn^T directly into d_out (WMMA, M=512 N=100000 K=256)
  {
    dim3 g(((kV / 16) + 7) / 8, kB / 64);
    k_gemm_nt<<<g, 256, 0, stream>>>(sr16, bn16, out, kB, kV, kD);
  }
  // 13. masked dual log-softmax, in place on d_out
  k_final<<<kB, 256, 0, stream>>>(out, iid, logphi);
}
